// RotaryAttention_8744553415084
// MI455X (gfx1250) — matrix operand order repaired and checked by reading, not compiled, not run on hardware
//
#include <hip/hip_runtime.h>
#include <hip/hip_bf16.h>

#define B_ 2
#define L_ 2048
#define D_ 1024
#define H_ 16
#define S_ 64
#define BL (B_ * L_)

typedef __bf16 bf16_t;
typedef __attribute__((ext_vector_type(16))) __bf16 v16bf;
typedef __attribute__((ext_vector_type(8)))  __bf16 v8bf;
typedef __attribute__((ext_vector_type(8)))  float  v8f;

__device__ __forceinline__ v8f vzero8() {
  v8f z;
#pragma unroll
  for (int i = 0; i < 8; ++i) z[i] = 0.0f;
  return z;
}

// 16-bit A-matrix 16x32 fragment (ISA 7.12.2): lane m = l&15;
// lanes 0-15 hold K={0..7,16..23}, lanes 16-31 hold K={8..15,24..31}.
__device__ __forceinline__ v16bf load_a_frag(const bf16_t* p, int ld, int lane) {
  const int m  = lane & 15;
  const int hi = lane >> 4;
  const bf16_t* r = p + (size_t)m * ld + hi * 8;
  v8bf lo = *(const v8bf*)(r);
  v8bf hh = *(const v8bf*)(r + 16);
  v16bf o;
#pragma unroll
  for (int i = 0; i < 8; ++i) { o[i] = lo[i]; o[i + 8] = hh[i]; }
  return o;
}

// 16-bit B-matrix 32x16 fragment: lane col n = l&15; lanes 0-15 hold
// K={0..7,16..23}, lanes 16-31 hold K={8..15,24..31} (column-n K contiguous, stride ldn).
__device__ __forceinline__ v16bf load_b_frag(const bf16_t* p, int ldn, int lane) {
  const int n  = lane & 15;
  const int hi = lane >> 4;
  const bf16_t* r = p + (size_t)n * ldn + hi * 8;
  v8bf lo = *(const v8bf*)(r);
  v8bf hh = *(const v8bf*)(r + 16);
  v16bf o;
#pragma unroll
  for (int i = 0; i < 8; ++i) { o[i] = lo[i]; o[i + 8] = hh[i]; }
  return o;
}

__device__ __forceinline__ v8f wmma_bf16(v16bf a, v16bf b, v8f c) {
  return __builtin_amdgcn_wmma_f32_16x16x32_bf16(false, a, false, b, (short)0, c,
                                                 false, false);
}

// ---------------- prep kernels ----------------
__global__ void cvt_bf16_kernel(const float* __restrict__ in, bf16_t* __restrict__ out,
                                int n) {
  int i = blockIdx.x * blockDim.x + threadIdx.x;
  if (i < n) out[i] = (bf16_t)in[i];
}

__global__ void transpose_bf16_kernel(const float* __restrict__ W,
                                      bf16_t* __restrict__ Wt) {
  int i = blockIdx.x * blockDim.x + threadIdx.x;  // D_*D_ threads
  int k = i >> 10, n = i & (D_ - 1);
  Wt[(size_t)n * D_ + k] = (bf16_t)W[(size_t)k * D_ + n];
}

// ---------------- fused QKV projection GEMM (bf16 WMMA) ----------------
// One wave computes a 16(M) x 64(N) strip of X @ W + b.
// vmode 0: write [BL, D] bf16 linear; vmode 1: write V transposed [B,H,S,L].
__global__ void __launch_bounds__(32)
qkv_gemm_kernel(const bf16_t* __restrict__ A, const bf16_t* __restrict__ Wt,
                const float* __restrict__ bias, bf16_t* __restrict__ outLin,
                bf16_t* __restrict__ outVt, int vmode) {
  const int lane = threadIdx.x;
  const int wave = blockIdx.x;           // 256 m-tiles * 16 n-strips
  const int nt = wave & 15;
  const int mt = wave >> 4;
  const int m0 = mt * 16, n0 = nt * 64;
  const int nlane = lane & 15, hil = lane >> 4;
  const bf16_t* Ab = A + (size_t)m0 * D_;

  v8f acc[4];
#pragma unroll
  for (int t = 0; t < 4; ++t) acc[t] = vzero8();

  for (int kb = 0; kb < D_; kb += 32) {
    const v16bf a = load_a_frag(Ab + kb, D_, lane);
#pragma unroll
    for (int t = 0; t < 4; ++t) {
      const v16bf bfr = load_b_frag(Wt + (size_t)(n0 + t * 16) * D_ + kb, D_, lane);
      acc[t] = wmma_bf16(a, bfr, acc[t]);
    }
  }

  if (vmode == 0) {
#pragma unroll
    for (int t = 0; t < 4; ++t) {
      const float bz = bias[n0 + t * 16 + nlane];
#pragma unroll
      for (int v = 0; v < 8; ++v)
        outLin[(size_t)(m0 + v + 8 * hil) * D_ + n0 + t * 16 + nlane] =
            (bf16_t)(acc[t][v] + bz);
    }
  } else {
#pragma unroll
    for (int t = 0; t < 4; ++t) {
      const int n = n0 + t * 16 + nlane;
      const float bz = bias[n];
      const int h = n >> 6, s = n & (S_ - 1);
#pragma unroll
      for (int v = 0; v < 8; ++v) {
        const int row = m0 + v + 8 * hil;
        const int bb = row >> 11, l = row & (L_ - 1);
        outVt[(((size_t)bb * H_ + h) * S_ + s) * L_ + l] = (bf16_t)(acc[t][v] + bz);
      }
    }
  }
}

// ---------------- rotary embedding ----------------
// One thread per (b, l, h, pair). Writes rq/rk in [B,H,L,S] bf16.
__global__ void rotary_kernel(const bf16_t* __restrict__ qlin,
                              const bf16_t* __restrict__ klin,
                              const float* __restrict__ temporal,
                              bf16_t* __restrict__ rq, bf16_t* __restrict__ rk) {
  const int i = blockIdx.x * blockDim.x + threadIdx.x;  // B*L*H*(S/2)
  const int s2 = i & 31;
  const int h  = (i >> 5) & (H_ - 1);
  const int l  = (i >> 9) & (L_ - 1);
  const int b  = i >> 20;
  const float t  = temporal[b * L_ + l];
  const float fr = __expf(-((float)(2 * s2) * (1.0f / (float)S_)) * 9.210340371976184f);
  float sn, cs;
  __sincosf(t * fr, &sn, &cs);
  const size_t lin = ((size_t)b * L_ + l) * D_ + h * S_ + 2 * s2;
  const float q0 = (float)qlin[lin], q1 = (float)qlin[lin + 1];
  const float k0 = (float)klin[lin], k1 = (float)klin[lin + 1];
  const size_t ro = (((size_t)b * H_ + h) * L_ + l) * S_ + 2 * s2;
  rq[ro]     = (bf16_t)(q0 * cs - q1 * sn);
  rq[ro + 1] = (bf16_t)(q1 * cs + q0 * sn);
  rk[ro]     = (bf16_t)(k0 * cs - k1 * sn);
  rk[ro + 1] = (bf16_t)(k1 * cs + k0 * sn);
}

// ---------------- flash attention (bf16 WMMA, online softmax) ----------------
// One wave per (b, h, 16-query tile). 32 keys per step: 4 score WMMAs,
// online softmax, P->LDS->A-fragment, 4 PV WMMAs.
__global__ void __launch_bounds__(32)
attn_fa_kernel(const bf16_t* __restrict__ rq, const bf16_t* __restrict__ rk,
               const bf16_t* __restrict__ vt, const float* __restrict__ mask,
               float* __restrict__ out) {
  __shared__ __align__(16) bf16_t Pt[16 * 32];
  const int lane = threadIdx.x;
  const int wave = blockIdx.x;                 // B_*H_*(L_/16)
  const int qt = wave & (L_ / 16 - 1);
  const int h  = (wave >> 7) & (H_ - 1);
  const int b  = wave >> 11;
  const int q0 = qt * 16;
  const int nlane = lane & 15, hil = lane >> 4;

  const bf16_t* rq_base = rq + (((size_t)(b * H_ + h)) * L_ + q0) * S_;
  const bf16_t* rk_base = rk + ((size_t)(b * H_ + h)) * L_ * S_;
  const bf16_t* vt_base = vt + ((size_t)(b * H_ + h)) * S_ * L_;

  const v16bf qa0 = load_a_frag(rq_base, S_, lane);        // head dims 0..31
  const v16bf qa1 = load_a_frag(rq_base + 32, S_, lane);   // head dims 32..63

  v8f acc[4];
#pragma unroll
  for (int t = 0; t < 4; ++t) acc[t] = vzero8();
  float mrow[8], srow[8];
#pragma unroll
  for (int v = 0; v < 8; ++v) { mrow[v] = -3.0e38f; srow[v] = 0.0f; }

  const float scl = 0.125f;                                // 1/sqrt(64)
  for (int kb = 0; kb < L_; kb += 32) {
    if (kb + 32 < L_) {
      __builtin_prefetch(rk_base + (size_t)(kb + 32) * S_ + lane * 64, 0, 1);
      __builtin_prefetch(vt_base + (size_t)(lane * 2) * L_ + kb + 32, 0, 1);
    }
    // ---- scores: S = rq @ rk^T (two 16x16 key tiles) ----
    const bf16_t* kp = rk_base + (size_t)kb * S_;
    v8f s0 = vzero8(), s1 = vzero8();
    s0 = wmma_bf16(qa0, load_b_frag(kp, S_, lane), s0);
    s0 = wmma_bf16(qa1, load_b_frag(kp + 32, S_, lane), s0);
    s1 = wmma_bf16(qa0, load_b_frag(kp + 16 * S_, S_, lane), s1);
    s1 = wmma_bf16(qa1, load_b_frag(kp + 16 * S_ + 32, S_, lane), s1);

    const float* mp = mask + (size_t)(q0 + 8 * hil) * L_ + kb + nlane;
#pragma unroll
    for (int v = 0; v < 8; ++v) {
      s0[v] = s0[v] * scl + mp[(size_t)v * L_];
      s1[v] = s1[v] * scl + mp[(size_t)v * L_ + 16];
    }

    // ---- online softmax (row reductions across 16-lane C-layout halves) ----
#pragma unroll
    for (int v = 0; v < 8; ++v) {
      float mx = fmaxf(s0[v], s1[v]);
#pragma unroll
      for (int off = 1; off < 16; off <<= 1)
        mx = fmaxf(mx, __shfl_xor(mx, off, 32));
      const float mnew = fmaxf(mrow[v], mx);
      const float corr = __expf(mrow[v] - mnew);
      const float p0 = __expf(s0[v] - mnew);
      const float p1 = __expf(s1[v] - mnew);
      float rs = p0 + p1;
#pragma unroll
      for (int off = 1; off < 16; off <<= 1)
        rs += __shfl_xor(rs, off, 32);
      srow[v] = srow[v] * corr + rs;
      mrow[v] = mnew;
      s0[v] = p0; s1[v] = p1;
#pragma unroll
      for (int t = 0; t < 4; ++t) acc[t][v] *= corr;
    }

    // ---- P (16x32) through LDS to re-shape C-layout -> A-fragment ----
#pragma unroll
    for (int v = 0; v < 8; ++v) {
      Pt[(v + 8 * hil) * 32 + nlane]      = (bf16_t)s0[v];
      Pt[(v + 8 * hil) * 32 + 16 + nlane] = (bf16_t)s1[v];
    }
    asm volatile("s_wait_dscnt 0" ::: "memory");
    const v16bf pa = load_a_frag(&Pt[0], 32, lane);

    // ---- context: acc += P @ V (4 N-tiles over head dim) ----
#pragma unroll
    for (int t = 0; t < 4; ++t) {
      const v16bf vb = load_b_frag(vt_base + (size_t)(t * 16) * L_ + kb, L_, lane);
      acc[t] = wmma_bf16(pa, vb, acc[t]);
    }
    asm volatile("s_wait_dscnt 0" ::: "memory");  // loads done before next P store
  }

  float* outp = out + ((size_t)b * L_ + q0) * D_ + h * S_;
#pragma unroll
  for (int t = 0; t < 4; ++t)
#pragma unroll
    for (int v = 0; v < 8; ++v)
      outp[(size_t)(v + 8 * hil) * D_ + t * 16 + nlane] = acc[t][v] / srow[v];
}

// ---------------- host launch ----------------
extern "C" void kernel_launch(void* const* d_in, const int* in_sizes, int n_in,
                              void* d_out, int out_size, void* d_ws, size_t ws_size,
                              hipStream_t stream) {
  (void)in_sizes; (void)n_in; (void)out_size; (void)ws_size;
  const float* data     = (const float*)d_in[0];
  const float* temporal = (const float*)d_in[1];
  const float* mask     = (const float*)d_in[2];
  const float* Wq = (const float*)d_in[3];
  const float* bq = (const float*)d_in[4];
  const float* Wk = (const float*)d_in[5];
  const float* bk = (const float*)d_in[6];
  const float* Wv = (const float*)d_in[7];
  const float* bv = (const float*)d_in[8];
  float* out = (float*)d_out;

  char* ws = (char*)d_ws;
  size_t off = 0;
  auto alloc = [&](size_t bytes) -> char* {
    char* p = ws + off;
    off += (bytes + 255) & ~(size_t)255;
    return p;
  };
  bf16_t* dataB = (bf16_t*)alloc((size_t)BL * D_ * sizeof(bf16_t));
  bf16_t* WtQ   = (bf16_t*)alloc((size_t)D_ * D_ * sizeof(bf16_t));
  bf16_t* WtK   = (bf16_t*)alloc((size_t)D_ * D_ * sizeof(bf16_t));
  bf16_t* WtV   = (bf16_t*)alloc((size_t)D_ * D_ * sizeof(bf16_t));
  bf16_t* qlin  = (bf16_t*)alloc((size_t)BL * D_ * sizeof(bf16_t));
  bf16_t* klin  = (bf16_t*)alloc((size_t)BL * D_ * sizeof(bf16_t));
  bf16_t* vtb   = (bf16_t*)alloc((size_t)B_ * H_ * S_ * L_ * sizeof(bf16_t));
  bf16_t* rqb   = (bf16_t*)alloc((size_t)B_ * H_ * L_ * S_ * sizeof(bf16_t));
  bf16_t* rkb   = (bf16_t*)alloc((size_t)B_ * H_ * L_ * S_ * sizeof(bf16_t));

  cvt_bf16_kernel<<<(BL * D_) / 256, 256, 0, stream>>>(data, dataB, BL * D_);
  transpose_bf16_kernel<<<(D_ * D_) / 256, 256, 0, stream>>>(Wq, WtQ);
  transpose_bf16_kernel<<<(D_ * D_) / 256, 256, 0, stream>>>(Wk, WtK);
  transpose_bf16_kernel<<<(D_ * D_) / 256, 256, 0, stream>>>(Wv, WtV);

  qkv_gemm_kernel<<<(BL / 16) * (D_ / 64), 32, 0, stream>>>(dataB, WtQ, bq, qlin, nullptr, 0);
  qkv_gemm_kernel<<<(BL / 16) * (D_ / 64), 32, 0, stream>>>(dataB, WtK, bk, klin, nullptr, 0);
  qkv_gemm_kernel<<<(BL / 16) * (D_ / 64), 32, 0, stream>>>(dataB, WtV, bv, nullptr, vtb, 1);

  rotary_kernel<<<(BL * H_ * (S_ / 2)) / 256, 256, 0, stream>>>(qlin, klin, temporal,
                                                                rqb, rkb);

  attn_fa_kernel<<<B_ * H_ * (L_ / 16), 32, 0, stream>>>(rqb, rkb, vtb, mask, out);
}
